// MaskGRU_model_27865747816718
// MI455X (gfx1250) — matrix operand order repaired and checked by reading, not compiled, not run on hardware
//
#include <hip/hip_runtime.h>
#include <math.h>

// ---------------------------------------------------------------------------
// MaskGRU forward: embed+input-proj GEMM (WMMA bf16), persistent dual GRU scan
// (WMMA bf16 + fused gates + grid barrier + gi prefetch), tiny FC head.
// Sizes fixed per reference: B=128, L=1024, E=256, H=512, 3H=1536, V=50000, NL=17
// ---------------------------------------------------------------------------

typedef __bf16 bf16_t;
typedef __attribute__((ext_vector_type(16))) __bf16 v16bf;
typedef __attribute__((ext_vector_type(8)))  __bf16 v8bf;
typedef __attribute__((ext_vector_type(8)))  float  v8f;

union ABu { v16bf v; v8bf h[2]; };

#define BB   128
#define LL   1024
#define EE   256
#define HH   512
#define G3   1536
#define NLAB 17
#define SCAN_WGS 32      // 32 WGs x 16 columns = 512 hidden cols
#define LSTR 520         // LDS row stride in bf16 elems (padded: conflict-free)

static __device__ inline v8bf cvt8(const float* __restrict__ p) {
  v8bf r;
#pragma unroll
  for (int i = 0; i < 8; ++i) r[i] = (bf16_t)p[i];
  return r;
}

static __device__ inline float sigmoidf_(float x) {
  return 1.0f / (1.0f + __expf(-x));
}

// ---------------- grid-wide sense barrier (persistent scan kernel) ----------
static __device__ inline void grid_barrier(int* cnt, int* gen, int nwg) {
  __threadfence();
  __syncthreads();
  if (threadIdx.x == 0) {
    int g = __hip_atomic_load(gen, __ATOMIC_RELAXED, __HIP_MEMORY_SCOPE_AGENT);
    if (__hip_atomic_fetch_add(cnt, 1, __ATOMIC_ACQ_REL, __HIP_MEMORY_SCOPE_AGENT)
        == nwg - 1) {
      __hip_atomic_store(cnt, 0, __ATOMIC_RELAXED, __HIP_MEMORY_SCOPE_AGENT);
      __hip_atomic_fetch_add(gen, 1, __ATOMIC_RELEASE, __HIP_MEMORY_SCOPE_AGENT);
    } else {
      while (__hip_atomic_load(gen, __ATOMIC_ACQUIRE, __HIP_MEMORY_SCOPE_AGENT) == g)
        __builtin_amdgcn_s_sleep(2);
    }
  }
  __syncthreads();
  __threadfence();
}

__global__ void init_bar_kernel(int* bar) { bar[0] = 0; bar[1] = 0; }

// ---------------------------------------------------------------------------
// Kernel 1: gi[l*B+b, 0:1536] = emb[x[b,l]] @ w_ih^T + b_ih   (stored bf16)
// Block 256 = 8 waves (2 M x 4 N). WG tile 32(M) x 256(N). Wave tile 16 x 64.
// Grid: (131072/32, 1536/256) = (4096, 6)
// ---------------------------------------------------------------------------
__global__ __launch_bounds__(256, 2)
void embed_inproj_kernel(const int* __restrict__ x,
                         const float* __restrict__ emb,
                         const float* __restrict__ w_ih,
                         const float* __restrict__ b_ih,
                         bf16_t* __restrict__ gi) {
  const int tid   = threadIdx.x;
  const int lane  = tid & 31;
  const int wave  = tid >> 5;
  const int Mbase = blockIdx.x * 32 + (wave & 1) * 16;
  const int Nbase = blockIdx.y * 256 + (wave >> 1) * 64;
  const int hh    = lane >> 4;   // K-half select (this code's contiguous k = 16*hh + s)
  const int colL  = lane & 15;

  // A-operand row for this lane: gather the embedding row (fused lookup)
  const int mrow = Mbase + colL;            // m = l*B + b
  const int l    = mrow >> 7;
  const int b    = mrow & (BB - 1);
  const int tok  = x[b * LL + l];
  const float* __restrict__ arow = emb + (size_t)tok * EE;

  const float* __restrict__ w0 = w_ih + (size_t)(Nbase +  0 + colL) * EE;
  const float* __restrict__ w1 = w_ih + (size_t)(Nbase + 16 + colL) * EE;
  const float* __restrict__ w2 = w_ih + (size_t)(Nbase + 32 + colL) * EE;
  const float* __restrict__ w3 = w_ih + (size_t)(Nbase + 48 + colL) * EE;

  v8f acc0 = {0.f,0.f,0.f,0.f,0.f,0.f,0.f,0.f};
  v8f acc1 = acc0, acc2 = acc0, acc3 = acc0;

#pragma unroll
  for (int kb = 0; kb < EE; kb += 32) {
    ABu ua;                                     // A 16x32: 16 contiguous K
    ua.h[0] = cvt8(arow + kb + hh * 16);
    ua.h[1] = cvt8(arow + kb + hh * 16 + 8);
    ABu ub;                                     // B 32x16: 16 contiguous K
    ub.h[0] = cvt8(w0 + kb + hh * 16);
    ub.h[1] = cvt8(w0 + kb + hh * 16 + 8);
    acc0 = __builtin_amdgcn_wmma_f32_16x16x32_bf16(false, ua.v, false, ub.v,
                                                   (short)0, acc0, false, false);
    ub.h[0] = cvt8(w1 + kb + hh * 16);
    ub.h[1] = cvt8(w1 + kb + hh * 16 + 8);
    acc1 = __builtin_amdgcn_wmma_f32_16x16x32_bf16(false, ua.v, false, ub.v,
                                                   (short)0, acc1, false, false);
    ub.h[0] = cvt8(w2 + kb + hh * 16);
    ub.h[1] = cvt8(w2 + kb + hh * 16 + 8);
    acc2 = __builtin_amdgcn_wmma_f32_16x16x32_bf16(false, ua.v, false, ub.v,
                                                   (short)0, acc2, false, false);
    ub.h[0] = cvt8(w3 + kb + hh * 16);
    ub.h[1] = cvt8(w3 + kb + hh * 16 + 8);
    acc3 = __builtin_amdgcn_wmma_f32_16x16x32_bf16(false, ua.v, false, ub.v,
                                                   (short)0, acc3, false, false);
  }

  const int c0 = Nbase + colL;
  const float bi0 = b_ih[c0], bi1 = b_ih[c0 + 16],
              bi2 = b_ih[c0 + 32], bi3 = b_ih[c0 + 48];
#pragma unroll
  for (int i = 0; i < 8; ++i) {
    const int   mi = Mbase + i + 8 * hh;       // D layout: VGPR i -> M=i / i+8
    const size_t g = (size_t)mi * G3;
    gi[g + c0     ] = (bf16_t)(acc0[i] + bi0);
    gi[g + c0 + 16] = (bf16_t)(acc1[i] + bi1);
    gi[g + c0 + 32] = (bf16_t)(acc2[i] + bi2);
    gi[g + c0 + 48] = (bf16_t)(acc3[i] + bi3);
  }
}

// ---------------------------------------------------------------------------
// Kernel 2: persistent forward+backward GRU scan.
// 32 WGs, each owns 16 hidden columns (r/z/n rows of w_hh staged in LDS bf16).
// h double-buffered in global (f32 master + bf16 matmul copy, both L2-hot).
// gi streams from HBM: prefetch next step's slice at the top of each step.
// One grid barrier per step; 2048 steps total.
// ---------------------------------------------------------------------------
__global__ __launch_bounds__(256, 1)
void gru_scan_kernel(const bf16_t* __restrict__ gi,
                     const float*  __restrict__ w_hh,
                     const float*  __restrict__ b_hh,
                     float*        __restrict__ h32,   // [2][128][512]
                     bf16_t*       __restrict__ hbf,   // [2][128][512]
                     float*        __restrict__ hf,    // [128][512]
                     float*        __restrict__ hb,    // [128][512]
                     int*          __restrict__ bar) {
  __shared__ bf16_t wsl[48 * LSTR];                    // ~50 KB
  const int tid = threadIdx.x;
  const int jb  = blockIdx.x * 16;                     // this WG's column base
  int* cnt = bar;
  int* gen = bar + 1;

  // Stage w_hh rows {jb..jb+15} of gates r/z/n into LDS (f32 -> bf16).
  for (int idx = tid; idx < 48 * HH; idx += 256) {
    const int row  = idx >> 9;                         // 0..47
    const int k    = idx & (HH - 1);
    const int gate = row >> 4;                         // 0=r 1=z 2=n
    const int grow = gate * HH + jb + (row & 15);
    wsl[row * LSTR + k] = (bf16_t)w_hh[(size_t)grow * HH + k];
  }
  // h0 = 0 in ping buffer 0 (this WG's columns only).
  for (int idx = tid; idx < BB * 16; idx += 256) {
    const int m = idx >> 4, c = idx & 15;
    h32[m * HH + jb + c] = 0.0f;
    hbf[m * HH + jb + c] = (bf16_t)0.0f;
  }
  grid_barrier(cnt, gen, (int)gridDim.x);

  const int lane  = tid & 31;
  const int wave  = tid >> 5;
  const int Mbase = wave * 16;                         // 8 waves cover B=128
  const int hh    = lane >> 4;
  const int colL  = lane & 15;
  const int jcol  = jb + colL;

  const float bhr = b_hh[jcol];
  const float bhz = b_hh[HH + jcol];
  const float bhn = b_hh[2 * HH + jcol];

  const bf16_t* __restrict__ wR = wsl + ( 0 + colL) * LSTR;
  const bf16_t* __restrict__ wZ = wsl + (16 + colL) * LSTR;
  const bf16_t* __restrict__ wN = wsl + (32 + colL) * LSTR;

  for (int c = 0; c < 2 * LL; ++c) {
    const int t = (c < LL) ? c : (2 * LL - 1 - c);     // fwd then reversed
    const int p = c & 1;                               // read buf p, write 1-p
    const float*  __restrict__ hs32 = h32 + p * (BB * HH);
    const bf16_t* __restrict__ hsbf = hbf + p * (BB * HH);
    float*        __restrict__ hd32 = h32 + (1 - p) * (BB * HH);
    bf16_t*       __restrict__ hdbf = hbf + (1 - p) * (BB * HH);
    const bf16_t* __restrict__ arow = hsbf + (size_t)(Mbase + colL) * HH;

    // Prefetch next step's gi slice for this WG (gi > L2 -> HBM stream);
    // overlaps HBM latency with this step's WMMA chain.
    if (c + 1 < 2 * LL) {
      const int tn = (c + 1 < LL) ? (c + 1) : (2 * LL - 2 - c);
      const bf16_t* gnx = gi + (size_t)tn * (BB * G3) + jb;
      for (int q = tid; q < 3 * BB; q += 256) {        // 3 gates x 128 rows
        const int m = q & (BB - 1), gate = q >> 7;
        __builtin_prefetch(gnx + (size_t)m * G3 + gate * HH, 0, 3);
      }
    }

    // Hoist this step's gate inputs and h_old so they fly behind the WMMAs.
    float gr[8], gz[8], gn[8], ho[8];
    const bf16_t* __restrict__ gi_t = gi + (size_t)t * (BB * G3);
#pragma unroll
    for (int i = 0; i < 8; ++i) {
      const int   mi = Mbase + i + 8 * hh;
      const size_t g = (size_t)mi * G3;
      gr[i] = (float)gi_t[g + jcol];
      gz[i] = (float)gi_t[g + HH + jcol];
      gn[i] = (float)gi_t[g + 2 * HH + jcol];
      ho[i] = hs32[mi * HH + jcol];
    }

    v8f accR = {0.f,0.f,0.f,0.f,0.f,0.f,0.f,0.f};
    v8f accZ = accR, accN = accR;
#pragma unroll
    for (int kb = 0; kb < HH; kb += 32) {
      ABu ua;
      ua.h[0] = *(const v8bf*)(arow + kb + hh * 16);
      ua.h[1] = *(const v8bf*)(arow + kb + hh * 16 + 8);
      ABu ub;
      const bf16_t* r0 = wR + kb + hh * 16;
      ub.h[0] = *(const v8bf*)(r0);
      ub.h[1] = *(const v8bf*)(r0 + 8);
      accR = __builtin_amdgcn_wmma_f32_16x16x32_bf16(false, ua.v, false, ub.v,
                                                     (short)0, accR, false, false);
      const bf16_t* z0 = wZ + kb + hh * 16;
      ub.h[0] = *(const v8bf*)(z0);
      ub.h[1] = *(const v8bf*)(z0 + 8);
      accZ = __builtin_amdgcn_wmma_f32_16x16x32_bf16(false, ua.v, false, ub.v,
                                                     (short)0, accZ, false, false);
      const bf16_t* n0 = wN + kb + hh * 16;
      ub.h[0] = *(const v8bf*)(n0);
      ub.h[1] = *(const v8bf*)(n0 + 8);
      accN = __builtin_amdgcn_wmma_f32_16x16x32_bf16(false, ua.v, false, ub.v,
                                                     (short)0, accN, false, false);
    }

    // Fused gates on the D fragment: VGPR i -> row (Mbase + i + 8*hh), col jcol.
#pragma unroll
    for (int i = 0; i < 8; ++i) {
      const int   mi = Mbase + i + 8 * hh;
      const float r  = sigmoidf_(gr[i] + accR[i] + bhr);
      const float z  = sigmoidf_(gz[i] + accZ[i] + bhz);
      const float n  = tanhf(gn[i] + r * (accN[i] + bhn));
      const float hn = (1.0f - z) * n + z * ho[i];
      hd32[mi * HH + jcol] = hn;
      hdbf[mi * HH + jcol] = (bf16_t)hn;
    }

    grid_barrier(cnt, gen, (int)gridDim.x);

    if (c == LL - 1) {                                 // forward final -> hf
      for (int idx = tid; idx < BB * 16; idx += 256) { // buf0 (step 1023 wrote it)
        const int m = idx >> 4, cc = idx & 15;
        hf[m * HH + jb + cc] = h32[m * HH + jb + cc];
      }  // next write of buf0 is gated behind the c=1024 barrier -> race-free
    }
  }
  // backward final (step 2047 wrote buf0) -> hb
  for (int idx = tid; idx < BB * 16; idx += 256) {
    const int m = idx >> 4, cc = idx & 15;
    hb[m * HH + jb + cc] = h32[m * HH + jb + cc];
  }
}

// ---------------------------------------------------------------------------
// Kernel 3: out = sigmoid([hf|hb] @ fc_w^T + fc_b), 128x17 outputs (tiny)
// ---------------------------------------------------------------------------
__global__ void fc_head_kernel(const float* __restrict__ hf,
                               const float* __restrict__ hb,
                               const float* __restrict__ fc_w,
                               const float* __restrict__ fc_b,
                               float* __restrict__ out) {
  const int idx = blockIdx.x * blockDim.x + threadIdx.x;
  if (idx >= BB * NLAB) return;
  const int b = idx / NLAB, n = idx % NLAB;
  const float* __restrict__ w   = fc_w + (size_t)n * (2 * HH);
  const float* __restrict__ hfr = hf + (size_t)b * HH;
  const float* __restrict__ hbr = hb + (size_t)b * HH;
  float s = fc_b[n];
  for (int k = 0; k < HH; ++k) s += hfr[k] * w[k];
  for (int k = 0; k < HH; ++k) s += hbr[k] * w[HH + k];
  out[idx] = sigmoidf_(s);
}

// ---------------------------------------------------------------------------
extern "C" void kernel_launch(void* const* d_in, const int* in_sizes, int n_in,
                              void* d_out, int out_size, void* d_ws, size_t ws_size,
                              hipStream_t stream) {
  (void)in_sizes; (void)n_in; (void)out_size; (void)ws_size;
  const int*   x    = (const int*)  d_in[0];
  const float* emb  = (const float*)d_in[1];
  const float* w_ih = (const float*)d_in[2];
  const float* w_hh = (const float*)d_in[3];
  const float* b_ih = (const float*)d_in[4];
  const float* b_hh = (const float*)d_in[5];
  const float* fc_w = (const float*)d_in[6];
  const float* fc_b = (const float*)d_in[7];
  float* out = (float*)d_out;

  // Workspace carve-up (~404 MB total; all regions 256B-aligned).
  char* ws = (char*)d_ws;
  size_t o = 0;
  bf16_t* gi  = (bf16_t*)(ws + o); o += (size_t)LL * BB * G3 * sizeof(bf16_t);
  float*  h32 = (float*) (ws + o); o += (size_t)2 * BB * HH * sizeof(float);
  bf16_t* hbf = (bf16_t*)(ws + o); o += (size_t)2 * BB * HH * sizeof(bf16_t);
  float*  hf  = (float*) (ws + o); o += (size_t)BB * HH * sizeof(float);
  float*  hb  = (float*) (ws + o); o += (size_t)BB * HH * sizeof(float);
  int*    bar = (int*)   (ws + o); o += 256;

  init_bar_kernel<<<1, 1, 0, stream>>>(bar);
  embed_inproj_kernel<<<dim3((BB * LL) / 32, G3 / 256), 256, 0, stream>>>(
      x, emb, w_ih, b_ih, gi);
  gru_scan_kernel<<<SCAN_WGS, 256, 0, stream>>>(
      gi, w_hh, b_hh, h32, hbf, hf, hb, bar);
  fc_head_kernel<<<(BB * NLAB + 255) / 256, 256, 0, stream>>>(
      hf, hb, fc_w, fc_b, out);
}
